// Seq2Seq_80144089743486
// MI455X (gfx1250) — compile-verified, matrix operand order checked by reading, not run on hardware
//
#include <hip/hip_runtime.h>
#include <hip/hip_bf16.h>

// ---------------------------------------------------------------------------
// MI455X (gfx1250) LAS seq2seq forward.
//   * All GEMMs use V_WMMA_F32_16X16X32_BF16 (bf16 A/B, f32 accumulate).
//   * Input projections are big parallel WMMA GEMMs with vectorized fp32->bf16
//     staging and async global->LDS weight staging (ASYNCcnt-tracked).
//   * Recurrence runs as a persistent single-workgroup kernel with h/c/gates
//     resident in LDS (~208KB of the 320KB WGP LDS), Whh streamed from L2.
//   * Decoder: persistent 512-thread workgroup, 160 steps, WMMA cells +
//     VALU attention.
// ---------------------------------------------------------------------------

typedef __attribute__((ext_vector_type(16))) __bf16 v16bf;
typedef __attribute__((ext_vector_type(4)))  __bf16 v4bf;
typedef __attribute__((ext_vector_type(8)))  float  v8f;
typedef __attribute__((ext_vector_type(4)))  float  v4f;

__device__ __forceinline__ v8f wmma_bf16(v16bf a, v16bf b, v8f c) {
  return __builtin_amdgcn_wmma_f32_16x16x32_bf16(false, a, false, b,
                                                 (short)0, c, false, false);
}
__device__ __forceinline__ float sigf(float x) { return 1.f / (1.f + __expf(-x)); }
__device__ __forceinline__ v4bf cvt4(v4f v) {
  v4bf o;
  o[0] = (__bf16)v[0]; o[1] = (__bf16)v[1];
  o[2] = (__bf16)v[2]; o[3] = (__bf16)v[3];
  return o;
}

// ---------------------------------------------------------------------------
// Weight packing: fp32 (N x src_ld) slice -> bf16 (N x dst_ld) slice.
// ---------------------------------------------------------------------------
__global__ void k_pack_bf16(const float* __restrict__ src, __bf16* __restrict__ dst,
                            int N, int src_ld, int src_off, int Kc,
                            int dst_ld, int dst_off) {
  long long i = (long long)blockIdx.x * blockDim.x + threadIdx.x;
  if (i >= (long long)N * Kc) return;
  int k = (int)(i % Kc);
  int n = (int)(i / Kc);
  dst[(size_t)n * dst_ld + dst_off + k] =
      (__bf16)src[(size_t)n * src_ld + src_off + k];
}

__global__ void k_zero_bf16(__bf16* __restrict__ p, long long n) {
  long long i = (long long)blockIdx.x * blockDim.x + threadIdx.x;
  if (i < n) p[i] = (__bf16)0.f;
}

// ---------------------------------------------------------------------------
// Generic GEMM: C[M,N] = A[M,Ka](f32) * W[N,Kp](bf16)^T (+bias).
// Block = 128 threads (4 waves), tile 32x64, K chunked by 32 through LDS.
// A staged as float4 -> packed bf16 (Ka%4==0 guaranteed); W staged with
// GLOBAL_LOAD_ASYNC_TO_LDS_B128 (ASYNCcnt), waited before the barrier.
// Requires M%32==0, N%64==0, Kp%32==0, Kp>=Ka, Ka%4==0.
// ---------------------------------------------------------------------------
__global__ __launch_bounds__(128) void k_gemm(const float* __restrict__ A,
                                              const __bf16* __restrict__ W,
                                              const float* __restrict__ bias,
                                              float* __restrict__ C,
                                              int M, int N, int Ka, int Kp) {
  __shared__ __bf16 As[32 * 32];
  __shared__ __bf16 Bs[64 * 32];
  const int m0 = blockIdx.y * 32;
  const int n0 = blockIdx.x * 64;
  const int tid = threadIdx.x;
  const int lane = tid & 31, wave = tid >> 5;
  const int l15 = lane & 15, l16 = lane >> 4;
  v8f acc0 = {}; v8f acc1 = {};
  for (int k0 = 0; k0 < Kp; k0 += 32) {
    // async-stage W tile: 64x32 bf16 = 4KB, 16B per lane-op, 2 ops/thread
    for (int g = tid; g < 256; g += 128) {
      int r = g >> 2, k8 = (g & 3) * 8;
      const __bf16* src = &W[(size_t)(n0 + r) * Kp + k0 + k8];
      unsigned lds = (unsigned)(size_t)&Bs[r * 32 + k8];
      asm volatile("global_load_async_to_lds_b128 %0, %1, off"
                   :: "v"(lds), "v"((unsigned long long)(size_t)src)
                   : "memory");
    }
    // stage A tile: 32x32, float4 loads, packed bf16 converts (no per-elem wait)
    for (int g = tid; g < 256; g += 128) {
      int r = g >> 3, k4 = (g & 7) * 4;
      int gk = k0 + k4;
      v4f av = {};
      if (gk < Ka) av = *(const v4f*)&A[(size_t)(m0 + r) * Ka + gk];
      *(v4bf*)&As[r * 32 + k4] = cvt4(av);
    }
    if (k0 + 32 < Kp)  // pull next weight chunk toward the WGP
      __builtin_prefetch(&W[(size_t)(n0 + (tid & 63)) * Kp + k0 + 32], 0, 1);
    asm volatile("s_wait_asynccnt 0x0" ::: "memory");
    __syncthreads();
    v16bf a0 = *(const v16bf*)&As[(l15) * 32 + l16 * 16];
    v16bf a1 = *(const v16bf*)&As[(16 + l15) * 32 + l16 * 16];
    v16bf b  = *(const v16bf*)&Bs[(wave * 16 + l15) * 32 + l16 * 16];
    acc0 = wmma_bf16(a0, b, acc0);
    acc1 = wmma_bf16(a1, b, acc1);
    __syncthreads();
  }
  const int n = n0 + wave * 16 + l15;
  const float bv = bias ? bias[n] : 0.f;
  for (int v = 0; v < 8; ++v) {
    int r0 = v + 8 * l16;  // documented 16x16 f32 C/D layout
    C[(size_t)(m0 + r0) * N + n]      = acc0[v] + bv;
    C[(size_t)(m0 + 16 + r0) * N + n] = acc1[v] + bv;
  }
}

// ---------------------------------------------------------------------------
// Per-sequence time reversal (t < len -> len-1-t), with stride/offset so the
// backward direction can be written straight into the concat buffer.
// ---------------------------------------------------------------------------
__global__ void k_reverse(const float* __restrict__ in, float* __restrict__ out,
                          const int* __restrict__ xlen, int shift,
                          int T, int D, int in_ld, int out_ld, int out_off) {
  long long i = (long long)blockIdx.x * blockDim.x + threadIdx.x;
  if (i >= (long long)32 * T * D) return;
  int d = (int)(i % D);
  long long bt = i / D;
  int t = (int)(bt % T);
  int b = (int)(bt / T);
  int len = xlen[b] >> shift;
  int ts = (t < len) ? (len - 1 - t) : t;
  out[((size_t)b * T + t) * out_ld + out_off + d] =
      in[((size_t)b * T + ts) * in_ld + d];
}

// ---------------------------------------------------------------------------
// Persistent masked LSTM recurrence. One workgroup of 1024 threads (32 waves).
// G = precomputed x@Wih^T (B*T,1024). Each step: h(bf16,LDS) @ Whh^T via
// 128 WMMA tiles (4/wave), gates in LDS, pointwise cell update in fp32.
// LDS: h(32K) c(32K) gates(128K) h_bf16(16K) = 208KB.
// ---------------------------------------------------------------------------
__global__ __launch_bounds__(1024) void k_lstm_seq(
    const float* __restrict__ G, const __bf16* __restrict__ Whh,
    const float* __restrict__ bih, const float* __restrict__ bhh,
    const int* __restrict__ xlen, int shift,
    float* __restrict__ Out, int out_ld, int out_off, int T) {
  extern __shared__ char smem_raw[];
  float* hs  = (float*)smem_raw;      // 32*256
  float* cs  = hs + 32 * 256;         // 32*256
  float* gsm = cs + 32 * 256;         // 32*1024
  __bf16* hbf = (__bf16*)(gsm + 32 * 1024);  // 32*256
  const int tid = threadIdx.x;
  const int lane = tid & 31, wave = tid >> 5;
  const int l15 = lane & 15, l16 = lane >> 4;
  for (int i = tid; i < 32 * 256; i += 1024) { hs[i] = 0.f; cs[i] = 0.f; }
  __syncthreads();
  for (int t = 0; t < T; ++t) {
    for (int g = tid; g < 2048; g += 1024)  // vectorized h -> bf16 restage
      *(v4bf*)&hbf[g * 4] = cvt4(*(const v4f*)&hs[g * 4]);
    __syncthreads();
    // h @ Whh^T : 2 Mtiles x 64 Ntiles, K=256 (8 chunks of 32)
    for (int j = 0; j < 4; ++j) {
      int id = wave * 4 + j;
      int nt = id >> 1, mt = id & 1;
      v8f acc = {};
      const __bf16* ar = hbf + (mt * 16 + l15) * 256 + l16 * 16;
      const __bf16* br = Whh + (size_t)(nt * 16 + l15) * 256 + l16 * 16;
      __builtin_prefetch(br + 256, 0, 1);
      for (int kk = 0; kk < 8; ++kk)
        acc = wmma_bf16(*(const v16bf*)(ar + kk * 32),
                        *(const v16bf*)(br + kk * 32), acc);
      int n = nt * 16 + l15;
      float bs = bih[n] + bhh[n];
      for (int v = 0; v < 8; ++v) {
        int b = mt * 16 + v + 8 * l16;
        gsm[b * 1024 + n] = G[((size_t)b * T + t) * 1024 + n] + acc[v] + bs;
      }
    }
    __syncthreads();
    // pointwise cell (gate order i,f,g,o), masked by t < len
    for (int k = 0; k < 8; ++k) {
      int i = tid + k * 1024;
      int b = i >> 8, j = i & 255;
      float gi = gsm[b * 1024 + j];
      float gf = gsm[b * 1024 + 256 + j];
      float gg = gsm[b * 1024 + 512 + j];
      float go = gsm[b * 1024 + 768 + j];
      float cn = sigf(gf) * cs[i] + sigf(gi) * tanhf(gg);
      float hn = sigf(go) * tanhf(cn);
      bool valid = t < (xlen[b] >> shift);
      if (valid) { cs[i] = cn; hs[i] = hn; }
      Out[((size_t)b * T + t) * out_ld + out_off + j] = valid ? hn : 0.f;
    }
    __syncthreads();
  }
}

// ---------------------------------------------------------------------------
// tf_in embedding (teacher forcing): tf[b,0]=E[SOS], tf[b,i]=E[y[b,i-1]]
// ---------------------------------------------------------------------------
__global__ void k_embed(const float* __restrict__ E, const int* __restrict__ y,
                        float* __restrict__ tf) {
  long long i = (long long)blockIdx.x * blockDim.x + threadIdx.x;
  if (i >= (long long)32 * 160 * 256) return;
  int d = (int)(i & 255);
  int bt = (int)(i >> 8);
  int t = bt % 160, b = bt / 160;
  int tok = (t == 0) ? 0 : y[b * 160 + t - 1];
  tf[i] = E[tok * 256 + d];
}

__global__ void k_ctx0(const float* __restrict__ Vp, float* __restrict__ ctx0) {
  int i = blockIdx.x * blockDim.x + threadIdx.x;
  if (i >= 32 * 128) return;
  int b = i >> 7, d = i & 127;
  float s = 0.f;
  for (int q = 0; q < 96; ++q) s += Vp[((size_t)b * 96 + q) * 128 + d];
  ctx0[i] = s * (1.f / 96.f);
}

// ---------------------------------------------------------------------------
// Persistent attention decoder: one workgroup, 512 threads (16 waves),
// 160 steps. d1/d2 cells via WMMA against pre-packed concat weights,
// attention in VALU, output projection via WMMA (E padded to 32 rows).
// LDS: h1/c1 64K, h2/c2/ctx 48K, gates 128K, att 12K, bf16 stage 24K = 276KB.
// ---------------------------------------------------------------------------
#define DEC_T 512
__global__ __launch_bounds__(DEC_T) void k_decoder(
    const float* __restrict__ pre1,   // (B,160,1024) = tf @ Wih[:, :256]^T
    const __bf16* __restrict__ Wd1,   // (1024,384)  [Wih[:,256:384] | Whh]
    const float* __restrict__ b1i, const float* __restrict__ b1h,
    const __bf16* __restrict__ Wd2,   // (512,384)   [Wih | Whh]
    const float* __restrict__ b2i, const float* __restrict__ b2h,
    const float* __restrict__ Kp, const float* __restrict__ Vp,  // (B,96,128)
    const float* __restrict__ ctx0, const __bf16* __restrict__ Ebf,  // (32,256)
    const float* __restrict__ b_out, const int* __restrict__ xlen,
    float* __restrict__ preds,        // (B,160,30)
    float* __restrict__ atts) {       // (160,96)
  extern __shared__ char smem_raw[];
  float* h1  = (float*)smem_raw;  // 8192
  float* c1  = h1 + 8192;         // 8192
  float* h2  = c1 + 8192;         // 4096
  float* c2  = h2 + 4096;         // 4096
  float* ctx = c2 + 4096;         // 4096
  float* gsm = ctx + 4096;        // 32768
  float* att = gsm + 32768;       // 3072
  __bf16* abuf = (__bf16*)(att + 3072);  // 32*384
  const int tid = threadIdx.x;
  const int lane = tid & 31, wave = tid >> 5;
  const int l15 = lane & 15, l16 = lane >> 4;
  for (int i = tid; i < 8192; i += DEC_T) { h1[i] = 0.f; c1[i] = 0.f; }
  for (int i = tid; i < 4096; i += DEC_T) { h2[i] = 0.f; c2[i] = 0.f; ctx[i] = ctx0[i]; }
  __syncthreads();
  const float scale = 0.08838834764831845f;  // 1/sqrt(128)
  for (int t = 0; t < 160; ++t) {
    // ---- stage A1 = [ctx | h1] (32 x 384), 4-wide (128-boundary is 4-aligned)
    for (int g = tid; g < 32 * 96; g += DEC_T) {
      int i4 = g * 4;
      int b = i4 / 384, d = i4 - b * 384;
      v4f v = (d < 128) ? *(const v4f*)&ctx[b * 128 + d]
                        : *(const v4f*)&h1[b * 256 + d - 128];
      *(v4bf*)&abuf[i4] = cvt4(v);
    }
    __syncthreads();
    // ---- d1 gates: 32x1024, K=384 (128 tiles / 16 waves) ----
    for (int j = 0; j < 8; ++j) {
      int id = wave * 8 + j;
      int nt = id >> 1, mt = id & 1;
      v8f acc = {};
      const __bf16* ar = abuf + (mt * 16 + l15) * 384 + l16 * 16;
      const __bf16* br = Wd1 + (size_t)(nt * 16 + l15) * 384 + l16 * 16;
      for (int kk = 0; kk < 12; ++kk)
        acc = wmma_bf16(*(const v16bf*)(ar + kk * 32),
                        *(const v16bf*)(br + kk * 32), acc);
      int n = nt * 16 + l15;
      float bs = b1i[n] + b1h[n];
      for (int v = 0; v < 8; ++v) {
        int b = mt * 16 + v + 8 * l16;
        gsm[b * 1024 + n] = pre1[((size_t)b * 160 + t) * 1024 + n] + acc[v] + bs;
      }
    }
    __syncthreads();
    for (int k = 0; k < 16; ++k) {  // d1 pointwise
      int i = tid + k * DEC_T;
      int b = i >> 8, j = i & 255;
      float gi = gsm[b * 1024 + j], gf = gsm[b * 1024 + 256 + j];
      float gg = gsm[b * 1024 + 512 + j], go = gsm[b * 1024 + 768 + j];
      float cn = sigf(gf) * c1[i] + sigf(gi) * tanhf(gg);
      c1[i] = cn; h1[i] = sigf(go) * tanhf(cn);
    }
    __syncthreads();
    // ---- stage A2 = [h1 | h2] ----
    for (int g = tid; g < 32 * 96; g += DEC_T) {
      int i4 = g * 4;
      int b = i4 / 384, d = i4 - b * 384;
      v4f v = (d < 256) ? *(const v4f*)&h1[b * 256 + d]
                        : *(const v4f*)&h2[b * 128 + d - 256];
      *(v4bf*)&abuf[i4] = cvt4(v);
    }
    __syncthreads();
    // ---- d2 gates: 32x512, K=384 (64 tiles / 16 waves) ----
    for (int j = 0; j < 4; ++j) {
      int id = wave * 4 + j;
      int nt = id >> 1, mt = id & 1;
      v8f acc = {};
      const __bf16* ar = abuf + (mt * 16 + l15) * 384 + l16 * 16;
      const __bf16* br = Wd2 + (size_t)(nt * 16 + l15) * 384 + l16 * 16;
      for (int kk = 0; kk < 12; ++kk)
        acc = wmma_bf16(*(const v16bf*)(ar + kk * 32),
                        *(const v16bf*)(br + kk * 32), acc);
      int n = nt * 16 + l15;
      float bs = b2i[n] + b2h[n];
      for (int v = 0; v < 8; ++v) {
        int b = mt * 16 + v + 8 * l16;
        gsm[b * 512 + n] = acc[v] + bs;
      }
    }
    __syncthreads();
    for (int k = 0; k < 8; ++k) {  // d2 pointwise
      int i = tid + k * DEC_T;
      int b = i >> 7, j = i & 127;
      float gi = gsm[b * 512 + j], gf = gsm[b * 512 + 128 + j];
      float gg = gsm[b * 512 + 256 + j], go = gsm[b * 512 + 384 + j];
      float cn = sigf(gf) * c2[i] + sigf(gi) * tanhf(gg);
      c2[i] = cn; h2[i] = sigf(go) * tanhf(cn);
    }
    __syncthreads();
    // ---- attention energies (per-batch GEMV, VALU) ----
    for (int k = 0; k < 6; ++k) {
      int i = tid + k * DEC_T;
      int b = i / 96, tq = i - b * 96;
      const float* kr = Kp + ((size_t)b * 96 + tq) * 128;
      float e = 0.f;
      for (int d = 0; d < 128; ++d) e += h2[b * 128 + d] * kr[d];
      e *= scale;
      if (tq >= (xlen[b] >> 3)) e = -1e9f;
      att[i] = e;
    }
    __syncthreads();
    if (tid < 32) {  // softmax, one row per thread
      float mx = -1e30f;
      for (int q = 0; q < 96; ++q) mx = fmaxf(mx, att[tid * 96 + q]);
      float s = 0.f;
      for (int q = 0; q < 96; ++q) {
        float e = __expf(att[tid * 96 + q] - mx);
        att[tid * 96 + q] = e; s += e;
      }
      float inv = 1.f / s;
      for (int q = 0; q < 96; ++q) att[tid * 96 + q] *= inv;
    }
    __syncthreads();
    if (tid < 96) atts[t * 96 + tid] = att[tid];  // att[0] row
    for (int k = 0; k < 8; ++k) {  // ctx = att @ V
      int i = tid + k * DEC_T;
      int b = i >> 7, d = i & 127;
      const float* vb = Vp + (size_t)b * 96 * 128 + d;
      float s = 0.f;
      for (int q = 0; q < 96; ++q) s += att[b * 96 + q] * vb[q * 128];
      ctx[i] = s;
    }
    __syncthreads();
    // ---- stage A3 = [h2 | ctx] (32 x 256) ----
    for (int g = tid; g < 32 * 64; g += DEC_T) {
      int i4 = g * 4;
      int b = i4 >> 8, d = i4 & 255;
      v4f v = (d < 128) ? *(const v4f*)&h2[b * 128 + d]
                        : *(const v4f*)&ctx[b * 128 + d - 128];
      *(v4bf*)&abuf[i4] = cvt4(v);
    }
    __syncthreads();
    // ---- pred = [h2|ctx] @ E^T + b_out (N padded 30->32) ----
    if (wave < 4) {
      int nt = wave >> 1, mt = wave & 1;
      v8f acc = {};
      const __bf16* ar = abuf + (mt * 16 + l15) * 256 + l16 * 16;
      const __bf16* br = Ebf + (nt * 16 + l15) * 256 + l16 * 16;
      for (int kk = 0; kk < 8; ++kk)
        acc = wmma_bf16(*(const v16bf*)(ar + kk * 32),
                        *(const v16bf*)(br + kk * 32), acc);
      int n = nt * 16 + l15;
      if (n < 30) {
        for (int v = 0; v < 8; ++v) {
          int b = mt * 16 + v + 8 * l16;
          preds[((size_t)b * 160 + t) * 30 + n] = acc[v] + b_out[n];
        }
      }
    }
    __syncthreads();
  }
}

// ---------------------------------------------------------------------------
// Host orchestration
// ---------------------------------------------------------------------------
static const size_t LSTM_SMEM = (size_t)(8192 + 8192 + 32768) * 4 + 8192 * 2;
static const size_t DEC_SMEM =
    (size_t)(8192 + 8192 + 4096 + 4096 + 4096 + 32768 + 3072) * 4 + 12288 * 2;

extern "C" void kernel_launch(void* const* d_in, const int* in_sizes, int n_in,
                              void* d_out, int out_size, void* d_ws, size_t ws_size,
                              hipStream_t stream) {
  (void)in_sizes; (void)n_in; (void)out_size; (void)ws_size;
  const float* x    = (const float*)d_in[0];
  const int*   xlen = (const int*)d_in[1];
  const int*   y    = (const int*)d_in[2];
  const float* eW[8][4];  // [layer*2+dir][Wih,Whh,bih,bhh]
  for (int l = 0; l < 8; ++l)
    for (int j = 0; j < 4; ++j) eW[l][j] = (const float*)d_in[3 + l * 4 + j];
  const float* Wk    = (const float*)d_in[35];
  const float* bk    = (const float*)d_in[36];
  const float* Wv    = (const float*)d_in[37];
  const float* bv    = (const float*)d_in[38];
  const float* E     = (const float*)d_in[39];
  const float* d1Wih = (const float*)d_in[40];
  const float* d1Whh = (const float*)d_in[41];
  const float* d1bih = (const float*)d_in[42];
  const float* d1bhh = (const float*)d_in[43];
  const float* d2Wih = (const float*)d_in[44];
  const float* d2Whh = (const float*)d_in[45];
  const float* d2bih = (const float*)d_in[46];
  const float* d2bhh = (const float*)d_in[47];
  const float* b_out = (const float*)d_in[48];

  // Bump allocator over d_ws.
  char* ws = (char*)d_ws;
  size_t off = 0;
  auto alloc = [&](size_t bytes) {
    void* p = ws + off;
    off = (off + bytes + 255) & ~(size_t)255;
    return p;
  };
  float* gates = (float*)alloc(sizeof(float) * (size_t)32 * 768 * 1024);
  float* revb  = (float*)alloc(sizeof(float) * (size_t)32 * 384 * 1024);
  float* catA  = (float*)alloc(sizeof(float) * (size_t)32 * 768 * 512);
  float* catB  = (float*)alloc(sizeof(float) * (size_t)32 * 768 * 512);
  float* tmpb  = (float*)alloc(sizeof(float) * (size_t)32 * 768 * 256);
  float* Kbuf  = (float*)alloc(sizeof(float) * 32 * 96 * 128);
  float* Vbuf  = (float*)alloc(sizeof(float) * 32 * 96 * 128);
  float* ctx0  = (float*)alloc(sizeof(float) * 32 * 128);
  float* tf    = (float*)alloc(sizeof(float) * (size_t)32 * 160 * 256);
  float* pre1  = (float*)alloc(sizeof(float) * (size_t)32 * 160 * 1024);
  __bf16* wih_bf[8];
  __bf16* whh_bf[8];
  for (int l = 0; l < 8; ++l) {
    wih_bf[l] = (__bf16*)alloc(2ull * 1024 * (l < 2 ? 96 : 1024));
    whh_bf[l] = (__bf16*)alloc(2ull * 1024 * 256);
  }
  __bf16* wd1emb = (__bf16*)alloc(2ull * 1024 * 256);
  __bf16* wd1cat = (__bf16*)alloc(2ull * 1024 * 384);
  __bf16* wd2cat = (__bf16*)alloc(2ull * 512 * 384);
  __bf16* wkbf   = (__bf16*)alloc(2ull * 128 * 512);
  __bf16* wvbf   = (__bf16*)alloc(2ull * 128 * 512);
  __bf16* ebf    = (__bf16*)alloc(2ull * 32 * 256);

  auto cdiv = [](long long a, long long b) { return (int)((a + b - 1) / b); };

  // ---- pack weights to bf16 (K padded to %32) ----
  k_zero_bf16<<<cdiv(1024 * 96, 256), 256, 0, stream>>>(wih_bf[0], 1024 * 96);
  k_zero_bf16<<<cdiv(1024 * 96, 256), 256, 0, stream>>>(wih_bf[1], 1024 * 96);
  k_zero_bf16<<<cdiv(32 * 256, 256), 256, 0, stream>>>(ebf, 32 * 256);
  for (int l = 0; l < 8; ++l) {
    int Din = (l < 2) ? 80 : 1024, Kpd = (l < 2) ? 96 : 1024;
    k_pack_bf16<<<cdiv((long long)1024 * Din, 256), 256, 0, stream>>>(
        eW[l][0], wih_bf[l], 1024, Din, 0, Din, Kpd, 0);
    k_pack_bf16<<<cdiv((long long)1024 * 256, 256), 256, 0, stream>>>(
        eW[l][1], whh_bf[l], 1024, 256, 0, 256, 256, 0);
  }
  k_pack_bf16<<<cdiv(1024 * 256, 256), 256, 0, stream>>>(d1Wih, wd1emb, 1024, 384, 0, 256, 256, 0);
  k_pack_bf16<<<cdiv(1024 * 128, 256), 256, 0, stream>>>(d1Wih, wd1cat, 1024, 384, 256, 128, 384, 0);
  k_pack_bf16<<<cdiv(1024 * 256, 256), 256, 0, stream>>>(d1Whh, wd1cat, 1024, 256, 0, 256, 384, 128);
  k_pack_bf16<<<cdiv(512 * 256, 256), 256, 0, stream>>>(d2Wih, wd2cat, 512, 256, 0, 256, 384, 0);
  k_pack_bf16<<<cdiv(512 * 128, 256), 256, 0, stream>>>(d2Whh, wd2cat, 512, 128, 0, 128, 384, 256);
  k_pack_bf16<<<cdiv(128 * 512, 256), 256, 0, stream>>>(Wk, wkbf, 128, 512, 0, 512, 512, 0);
  k_pack_bf16<<<cdiv(128 * 512, 256), 256, 0, stream>>>(Wv, wvbf, 128, 512, 0, 512, 512, 0);
  k_pack_bf16<<<cdiv(30 * 256, 256), 256, 0, stream>>>(E, ebf, 30, 256, 0, 256, 256, 0);

  // ---- encoder: BiLSTM + 3 pBLSTMs (pooling = row-major reinterpret) ----
  const float* inp = x;
  float* cat_cur = catA;
  float* cat_nxt = catB;
  int T = 768;
  for (int li = 0; li < 4; ++li) {
    const int Din = (li == 0) ? 80 : 1024;
    const int Kpd = (li == 0) ? 96 : 1024;
    const int sh = li;               // lens = x_len >> li
    const int M = 32 * T;
    const int fi = li * 2, bi = li * 2 + 1;
    // forward dir
    k_gemm<<<dim3(16, M / 32), 128, 0, stream>>>(inp, wih_bf[fi], nullptr, gates,
                                                 M, 1024, Din, Kpd);
    k_lstm_seq<<<1, 1024, LSTM_SMEM, stream>>>(gates, whh_bf[fi], eW[fi][2],
                                               eW[fi][3], xlen, sh, cat_cur, 512, 0, T);
    // backward dir: reverse input, run, reverse output into concat cols 256..511
    k_reverse<<<cdiv((long long)M * Din, 256), 256, 0, stream>>>(
        inp, revb, xlen, sh, T, Din, Din, Din, 0);
    k_gemm<<<dim3(16, M / 32), 128, 0, stream>>>(revb, wih_bf[bi], nullptr, gates,
                                                 M, 1024, Din, Kpd);
    k_lstm_seq<<<1, 1024, LSTM_SMEM, stream>>>(gates, whh_bf[bi], eW[bi][2],
                                               eW[bi][3], xlen, sh, tmpb, 256, 0, T);
    k_reverse<<<cdiv((long long)M * 256, 256), 256, 0, stream>>>(
        tmpb, cat_cur, xlen, sh, T, 256, 256, 512, 256);
    inp = cat_cur;  // next layer views (B,T,512) as (B,T/2,1024)
    float* tswap = cat_cur; cat_cur = cat_nxt; cat_nxt = tswap;
    T /= 2;
  }
  // inp = encoder output (B, 96, 512), lens = x_len >> 3

  // ---- K / V projections ----
  k_gemm<<<dim3(2, 96), 128, 0, stream>>>(inp, wkbf, bk, Kbuf, 3072, 128, 512, 512);
  k_gemm<<<dim3(2, 96), 128, 0, stream>>>(inp, wvbf, bv, Vbuf, 3072, 128, 512, 512);
  k_ctx0<<<cdiv(32 * 128, 256), 256, 0, stream>>>(Vbuf, ctx0);

  // ---- decoder pre-gates: tf_in @ Wih_d1[:, :256]^T ----
  k_embed<<<cdiv((long long)32 * 160 * 256, 256), 256, 0, stream>>>(E, y, tf);
  k_gemm<<<dim3(16, 160), 128, 0, stream>>>(tf, wd1emb, nullptr, pre1,
                                            5120, 1024, 256, 256);

  // ---- decoder ----
  float* preds = (float*)d_out;
  float* atts = preds + (size_t)32 * 160 * 30;
  k_decoder<<<1, DEC_T, DEC_SMEM, stream>>>(pre1, wd1cat, d1bih, d1bhh, wd2cat,
                                            d2bih, d2bhh, Kbuf, Vbuf, ctx0, ebf,
                                            b_out, xlen, preds, atts);
}
